// Seq2seq_29892972380468
// MI455X (gfx1250) — matrix operand order repaired and checked by reading, not compiled, not run on hardware
//
#include <hip/hip_runtime.h>
#include <math.h>

typedef __attribute__((ext_vector_type(16))) __bf16 v16bf;
typedef __attribute__((ext_vector_type(8)))  __bf16 v8bf;
typedef __attribute__((ext_vector_type(8)))  float  v8f;

constexpr int kH    = 1024;
constexpr int k2H   = 2048;
constexpr int k3H   = 3072;
constexpr int kSeq  = 512;     // input/target length == MAX_LEN
constexpr int kV    = 32000;
constexpr int kNT   = kV / 64; // 500 column tiles of 64
constexpr int kVSpl = 8;       // vocab splits (grid parallelism)

// ---------------------------------------------------------------------------
// WMMA fragment helpers (CDNA5 16x16x32 bf16, wave32). See ISA 7.12.2 layouts.
// ---------------------------------------------------------------------------

// A-matrix 16x32 (MxK) bf16 fragment from a row-major panel (lda in elems).
// lane<16: row=lane, K chunks [k0+0..7],[k0+16..23]; lanes>=16: +8 K shift.
__device__ __forceinline__ v16bf load_a_frag(const __bf16* A, int lda, int k0, int lane) {
    int m  = lane & 15;
    int hi = lane >> 4;
    const __bf16* p = A + m * lda + k0 + 8 * hi;
    v8bf c0 = *(const v8bf*)(p);
    v8bf c1 = *(const v8bf*)(p + 16);
    v16bf r;
#pragma unroll
    for (int j = 0; j < 8; ++j) { r[j] = c0[j]; r[8 + j] = c1[j]; }
    return r;
}

// B-matrix 32x16 (KxN) bf16: column n = weight row (n0+n) (row base pre-applied).
// lane<16: K chunks [k0+0..7],[k0+16..23]; lanes>=16: +8 K shift (matches A).
__device__ __forceinline__ v16bf load_b_frag(const __bf16* W, int ldw, int k0, int lane) {
    int n  = lane & 15;
    int kh = lane >> 4;
    const __bf16* p = W + n * ldw + k0 + 8 * kh;
    v8bf c0 = *(const v8bf*)(p);
    v8bf c1 = *(const v8bf*)(p + 16);
    v16bf r;
#pragma unroll
    for (int j = 0; j < 8; ++j) { r[j] = c0[j]; r[8 + j] = c1[j]; }
    return r;
}

__device__ __forceinline__ v8f wmma_bf16(v16bf a, v16bf b, v8f c) {
    return __builtin_amdgcn_wmma_f32_16x16x32_bf16(false, a, false, b, (short)0, c,
                                                   false, false);
}

// Stage a contiguous bf16 panel from global into LDS with the CDNA5 async
// copy path (GLOBAL_LOAD_ASYNC_TO_LDS_B128, tracked by ASYNCcnt).
// Generic->LDS pointer truncation to 32 bits yields the LDS byte offset
// (ISA 10.2).
__device__ __forceinline__ void stage_panel_async(const __bf16* g, __bf16* lds_dst,
                                                  int elems, int tid, int nthreads) {
    for (int c = tid * 8; c < elems; c += nthreads * 8) {   // 16 B per chunk
        unsigned loff = (unsigned)(uintptr_t)(lds_dst + c);
        const __bf16* gp = g + c;
        asm volatile("global_load_async_to_lds_b128 %0, %1, off"
                     :: "v"(loff), "v"(gp) : "memory");
    }
    asm volatile("s_wait_asynccnt 0" ::: "memory");
}

// online-softmax merge: (m1,s1) <- (m1,s1) + (m2,s2)
__device__ __forceinline__ void osm_merge(float& m1, float& s1, float m2, float s2) {
    float nm = fmaxf(m1, m2);
    s1 = s1 * __expf(m1 - nm) + s2 * __expf(m2 - nm);
    m1 = nm;
}

// ---------------------------------------------------------------------------
// 0) fp32 -> bf16 weight conversion (one pass; removes cvt from WMMA loops)
// ---------------------------------------------------------------------------
__global__ __launch_bounds__(256)
void f32_to_bf16_kernel(const float* __restrict__ src, __bf16* __restrict__ dst) {
    int idx = (blockIdx.x * 256 + threadIdx.x) * 8;
    float4 a = *(const float4*)(src + idx);
    float4 b = *(const float4*)(src + idx + 4);
    v8bf o;
    o[0] = (__bf16)a.x; o[1] = (__bf16)a.y; o[2] = (__bf16)a.z; o[3] = (__bf16)a.w;
    o[4] = (__bf16)b.x; o[5] = (__bf16)b.y; o[6] = (__bf16)b.z; o[7] = (__bf16)b.w;
    *(v8bf*)(dst + idx) = o;
}

// ---------------------------------------------------------------------------
// 1) Gather encoder embeddings -> bf16 activation matrix X [512 x 1024]
// ---------------------------------------------------------------------------
__global__ __launch_bounds__(256)
void gather_emb_kernel(const int* __restrict__ input, const float* __restrict__ emb,
                       __bf16* __restrict__ Xb) {
    int idx = blockIdx.x * 256 + threadIdx.x;      // 512*1024 elems
    int t = idx >> 10, k = idx & (kH - 1);
    Xb[idx] = (__bf16)emb[input[t] * kH + k];
}

// ---------------------------------------------------------------------------
// 2) Gi = X @ enc_W_ih^T + b_ih   (M=512, N=3072, K=1024)  — WMMA GEMM
//    block = 8 waves, A panel (16x1024 bf16) staged to LDS, 16x64 tile/wave.
// ---------------------------------------------------------------------------
__global__ __launch_bounds__(256)
void gi_gemm_kernel(const __bf16* __restrict__ Xb, const __bf16* __restrict__ Wih,
                    const float* __restrict__ bih, float* __restrict__ Gi) {
    __shared__ __align__(16) __bf16 Ash[16 * kH];
    int tid = threadIdx.x, wave = tid >> 5, lane = tid & 31;
    int m0 = blockIdx.y * 16;
    int n0 = blockIdx.x * 512 + wave * 64;

    stage_panel_async(Xb + m0 * kH, Ash, 16 * kH, tid, 256);
    __syncthreads();

    v8f acc[4] = {};
    for (int k0 = 0; k0 < kH; k0 += 32) {
        v16bf a = load_a_frag(Ash, kH, k0, lane);
#pragma unroll
        for (int s = 0; s < 4; ++s) {
            v16bf b = load_b_frag(Wih + (n0 + 16 * s) * kH, kH, k0, lane);
            acc[s] = wmma_bf16(a, b, acc[s]);
        }
    }
    int nlo = lane & 15, hi = lane >> 4;
#pragma unroll
    for (int s = 0; s < 4; ++s) {
        int col = n0 + 16 * s + nlo;
        float bb = bih[col];
#pragma unroll
        for (int r = 0; r < 8; ++r) {
            int row = m0 + r + 8 * hi;
            Gi[row * k3H + col] = acc[s][r] + bb;
        }
    }
}

// ---------------------------------------------------------------------------
// 3) Encoder recurrence (persistent single workgroup, h LDS-resident)
// ---------------------------------------------------------------------------
__global__ __launch_bounds__(1024)
void enc_seq_kernel(const float* __restrict__ Whh, const float* __restrict__ bhh,
                    const float* __restrict__ Gi, float* __restrict__ encO) {
    __shared__ float h[kH];
    __shared__ float gh[k3H];
    int tid = threadIdx.x, wave = tid >> 5, lane = tid & 31;
    h[tid] = 0.f;
    __syncthreads();

    for (int t = 0; t < kSeq; ++t) {
        // gh = W_hh @ h + b_hh : wave-per-row, coalesced K reads, shfl reduce
        for (int r = wave; r < k3H; r += 32) {
            const float* wr = Whh + r * kH;
            if (r + 32 < k3H) __builtin_prefetch(Whh + (r + 32) * kH + lane * 32, 0, 0);
            float p = 0.f;
            for (int k = lane; k < kH; k += 32) p += wr[k] * h[k];
#pragma unroll
            for (int off = 16; off > 0; off >>= 1) p += __shfl_xor(p, off, 32);
            if (lane == 0) gh[r] = p + bhh[r];
        }
        __syncthreads();
        const float* git = Gi + t * k3H;
        float gir = git[tid], giz = git[kH + tid], gin = git[2 * kH + tid];
        float ghr = gh[tid], ghz = gh[kH + tid], ghn = gh[2 * kH + tid];
        float rg = 1.f / (1.f + __expf(-(gir + ghr)));
        float zg = 1.f / (1.f + __expf(-(giz + ghz)));
        float ng = tanhf(gin + rg * ghn);
        float hn = (1.f - zg) * ng + zg * h[tid];
        __syncthreads();
        h[tid] = hn;
        encO[t * kH + tid] = hn;
        __syncthreads();
    }
}

// ---------------------------------------------------------------------------
// 4) Decoder recurrence: attention + combine + GRU, persistent workgroup
// ---------------------------------------------------------------------------
__global__ __launch_bounds__(1024)
void dec_seq_kernel(const int* __restrict__ target, const float* __restrict__ dec_emb,
                    const float* __restrict__ attn_W, const float* __restrict__ attn_b,
                    const float* __restrict__ comb_W, const float* __restrict__ comb_b,
                    const float* __restrict__ Wih, const float* __restrict__ Whh,
                    const float* __restrict__ bih, const float* __restrict__ bhh,
                    const float* __restrict__ encO, __bf16* __restrict__ Hdec) {
    __shared__ float h[kH], e[kH], x[kH], applied[kH];
    __shared__ float aw[kSeq];
    __shared__ float gi[k3H], gh[k3H];
    __shared__ float red[32];
    int tid = threadIdx.x, wave = tid >> 5, lane = tid & 31;

    h[tid] = encO[(kSeq - 1) * kH + tid];          // h0 = encoder final state
    __syncthreads();

    for (int t = 0; t < kSeq; ++t) {
        int tok = (t == 0) ? 0 : target[t - 1];    // SOS = 0, teacher forcing
        e[tid] = dec_emb[tok * kH + tid];
        __syncthreads();

        // attention scores: attn_W @ [e;h] + attn_b  (512 rows of 2048-dot)
        for (int j = wave; j < kSeq; j += 32) {
            const float* wr = attn_W + j * k2H;
            float p = 0.f;
            for (int k = lane; k < k2H; k += 32) {
                float v = (k < kH) ? e[k] : h[k - kH];
                p += wr[k] * v;
            }
#pragma unroll
            for (int off = 16; off > 0; off >>= 1) p += __shfl_xor(p, off, 32);
            if (lane == 0) aw[j] = p + attn_b[j];
        }
        __syncthreads();

        // softmax over 512 (block reduction: wave shfl + LDS)
        float s = (tid < kSeq) ? aw[tid] : -3.402823466e38f;
        float m = s;
#pragma unroll
        for (int off = 16; off > 0; off >>= 1) m = fmaxf(m, __shfl_xor(m, off, 32));
        if (lane == 0) red[wave] = m;
        __syncthreads();
        if (wave == 0) {
            float mm = red[lane];
#pragma unroll
            for (int off = 16; off > 0; off >>= 1) mm = fmaxf(mm, __shfl_xor(mm, off, 32));
            if (lane == 0) red[0] = mm;
        }
        __syncthreads();
        float mx = red[0];
        float ex = (tid < kSeq) ? __expf(s - mx) : 0.f;
        float sm = ex;
#pragma unroll
        for (int off = 16; off > 0; off >>= 1) sm += __shfl_xor(sm, off, 32);
        __syncthreads();
        if (lane == 0) red[wave] = sm;
        __syncthreads();
        if (wave == 0) {
            float ss = red[lane];
#pragma unroll
            for (int off = 16; off > 0; off >>= 1) ss += __shfl_xor(ss, off, 32);
            if (lane == 0) red[0] = ss;
        }
        __syncthreads();
        float inv = 1.f / red[0];
        if (tid < kSeq) aw[tid] = ex * inv;
        __syncthreads();

        // applied = aw @ enc_full  (column reads: lane-contiguous -> coalesced)
        float ap = 0.f;
        for (int j = 0; j < kSeq; ++j) ap += aw[j] * encO[j * kH + tid];
        applied[tid] = ap;
        __syncthreads();

        // x = relu(comb_W @ [e;applied] + comb_b)
        for (int r = wave; r < kH; r += 32) {
            const float* wr = comb_W + r * k2H;
            float p = 0.f;
            for (int k = lane; k < k2H; k += 32) {
                float v = (k < kH) ? e[k] : applied[k - kH];
                p += wr[k] * v;
            }
#pragma unroll
            for (int off = 16; off > 0; off >>= 1) p += __shfl_xor(p, off, 32);
            if (lane == 0) x[r] = fmaxf(p + comb_b[r], 0.f);
        }
        __syncthreads();

        // GRU: gi = W_ih@x + b_ih ; gh = W_hh@h + b_hh  (6144 row-jobs)
        for (int job = wave; job < 2 * k3H; job += 32) {
            int isGh = job >= k3H;
            int r = isGh ? job - k3H : job;
            const float* wr = (isGh ? Whh : Wih) + r * kH;
            const float* vv = isGh ? h : x;
            if (job + 32 < 2 * k3H) {
                int j2 = job + 32;
                const float* pf = (j2 >= k3H ? Whh + (j2 - k3H) * kH : Wih + j2 * kH);
                __builtin_prefetch(pf + lane * 32, 0, 0);
            }
            float p = 0.f;
            for (int k = lane; k < kH; k += 32) p += wr[k] * vv[k];
#pragma unroll
            for (int off = 16; off > 0; off >>= 1) p += __shfl_xor(p, off, 32);
            if (lane == 0) { if (isGh) gh[r] = p + bhh[r]; else gi[r] = p + bih[r]; }
        }
        __syncthreads();

        float gir = gi[tid], giz = gi[kH + tid], gin = gi[2 * kH + tid];
        float ghr = gh[tid], ghz = gh[kH + tid], ghn = gh[2 * kH + tid];
        float rg = 1.f / (1.f + __expf(-(gir + ghr)));
        float zg = 1.f / (1.f + __expf(-(giz + ghz)));
        float ng = tanhf(gin + rg * ghn);
        float hn = (1.f - zg) * ng + zg * h[tid];
        __syncthreads();
        h[tid] = hn;
        Hdec[t * kH + tid] = (__bf16)hn;
        __syncthreads();
    }
}

// ---------------------------------------------------------------------------
// 5) Output projection + online log-softmax partials (WMMA)
//    grid = (32 M-tiles) x (8 vocab splits) = 256 blocks for full occupancy.
//    Each block: 16 rows (A panel in LDS), 8 waves sweep a strided subset of
//    the 500 column tiles; emits per-(row,split) running (max, sumexp) and
//    the target logit if its column was seen.
// ---------------------------------------------------------------------------
__global__ __launch_bounds__(256)
void out_nll_partial_kernel(const __bf16* __restrict__ Hdec,
                            const __bf16* __restrict__ outW,
                            const float* __restrict__ outb,
                            const int* __restrict__ target,
                            float* __restrict__ pmax, float* __restrict__ psum,
                            float* __restrict__ ptgt) {
    __shared__ __align__(16) __bf16 Ash[16 * kH];   // 32 KB A panel
    __shared__ float wm[16][8];        // per-wave running max, per row
    __shared__ float wsum[16][8];      // per-wave running sum-exp, per row
    __shared__ float tgt_logit[16];
    int tid = threadIdx.x, wave = tid >> 5, lane = tid & 31;
    int nlo = lane & 15, hi = lane >> 4;
    int m0    = blockIdx.x * 16;
    int split = blockIdx.y;
    if (tid < 16) tgt_logit[tid] = -3.402823466e38f;

    stage_panel_async(Hdec + m0 * kH, Ash, 16 * kH, tid, 256);
    __syncthreads();

    int vt[8];
#pragma unroll
    for (int r = 0; r < 8; ++r) vt[r] = target[m0 + r + 8 * hi];
    float rm[8], rs[8];
#pragma unroll
    for (int r = 0; r < 8; ++r) { rm[r] = -3.402823466e38f; rs[r] = 0.f; }

    // global tile stream: (split, wave) pair picks tiles strided by 64
    for (int nt = split * 8 + wave; nt < kNT; nt += kVSpl * 8) {
        int n0 = nt * 64;
        v8f acc[4] = {};
        for (int k0 = 0; k0 < kH; k0 += 32) {
            v16bf a = load_a_frag(Ash, kH, k0, lane);
#pragma unroll
            for (int s = 0; s < 4; ++s) {
                v16bf b = load_b_frag(outW + (n0 + 16 * s) * kH, kH, k0, lane);
                acc[s] = wmma_bf16(a, b, acc[s]);
            }
        }
        // online softmax update for the 64 columns of this tile
#pragma unroll
        for (int r = 0; r < 8; ++r) {
            float val[4];
            float cm = -3.402823466e38f;
#pragma unroll
            for (int s = 0; s < 4; ++s) {
                int col = n0 + 16 * s + nlo;
                val[s] = acc[s][r] + outb[col];
                if (col == vt[r]) tgt_logit[r + 8 * hi] = val[s];
                cm = fmaxf(cm, val[s]);
            }
            float cs = 0.f;
#pragma unroll
            for (int s = 0; s < 4; ++s) cs += __expf(val[s] - cm);
            // merge across the 16 lanes that share this row
#pragma unroll
            for (int off = 1; off < 16; off <<= 1) {
                float om = __shfl_xor(cm, off, 32);
                float os = __shfl_xor(cs, off, 32);
                osm_merge(cm, cs, om, os);
            }
            osm_merge(rm[r], rs[r], cm, cs);
        }
    }
    if (nlo == 0) {
#pragma unroll
        for (int r = 0; r < 8; ++r) { wm[r + 8 * hi][wave] = rm[r]; wsum[r + 8 * hi][wave] = rs[r]; }
    }
    __syncthreads();
    if (tid < 16) {
        float fm = -3.402823466e38f, fs = 0.f;
#pragma unroll
        for (int w = 0; w < 8; ++w) osm_merge(fm, fs, wm[tid][w], wsum[tid][w]);
        int row = m0 + tid;
        pmax[row * kVSpl + split] = fm;
        psum[row * kVSpl + split] = fs;
        ptgt[row * kVSpl + split] = tgt_logit[tid];   // -inf if not in this split
    }
}

// ---------------------------------------------------------------------------
// 6) Merge vocab-split partials per row, then mean NLL
// ---------------------------------------------------------------------------
__global__ __launch_bounds__(512)
void nll_finalize_kernel(const float* __restrict__ pmax, const float* __restrict__ psum,
                         const float* __restrict__ ptgt, float* __restrict__ out) {
    __shared__ float red[16];
    int tid = threadIdx.x;                       // 512 threads = 512 rows
    float fm = -3.402823466e38f, fs = 0.f, tg = -3.402823466e38f;
#pragma unroll
    for (int s = 0; s < kVSpl; ++s) {
        osm_merge(fm, fs, pmax[tid * kVSpl + s], psum[tid * kVSpl + s]);
        tg = fmaxf(tg, ptgt[tid * kVSpl + s]);   // exactly one split saw it
    }
    float v = -(tg - fm - logf(fs));             // -log_softmax at target
#pragma unroll
    for (int off = 16; off > 0; off >>= 1) v += __shfl_xor(v, off, 32);
    if ((tid & 31) == 0) red[tid >> 5] = v;
    __syncthreads();
    if (tid < 16) {
        float s = red[tid];
#pragma unroll
        for (int off = 8; off > 0; off >>= 1) s += __shfl_xor(s, off, 32);
        if (tid == 0) out[0] = s / (float)kSeq;
    }
}

// ---------------------------------------------------------------------------
extern "C" void kernel_launch(void* const* d_in, const int* in_sizes, int n_in,
                              void* d_out, int out_size, void* d_ws, size_t ws_size,
                              hipStream_t stream) {
    const int*   input   = (const int*)d_in[0];
    const int*   target  = (const int*)d_in[1];
    const float* enc_emb = (const float*)d_in[2];
    const float* enc_Wih = (const float*)d_in[3];
    const float* enc_Whh = (const float*)d_in[4];
    const float* enc_bih = (const float*)d_in[5];
    const float* enc_bhh = (const float*)d_in[6];
    const float* dec_emb = (const float*)d_in[7];
    const float* attn_W  = (const float*)d_in[8];
    const float* attn_b  = (const float*)d_in[9];
    const float* comb_W  = (const float*)d_in[10];
    const float* comb_b  = (const float*)d_in[11];
    const float* dec_Wih = (const float*)d_in[12];
    const float* dec_Whh = (const float*)d_in[13];
    const float* dec_bih = (const float*)d_in[14];
    const float* dec_bhh = (const float*)d_in[15];
    const float* out_W   = (const float*)d_in[16];
    const float* out_b   = (const float*)d_in[17];
    float* out = (float*)d_out;

    char* ws = (char*)d_ws;
    __bf16* Xb    = (__bf16*)ws;  ws += (size_t)kSeq * kH * sizeof(__bf16);
    float*  Gi    = (float*)ws;   ws += (size_t)kSeq * k3H * sizeof(float);
    float*  encO  = (float*)ws;   ws += (size_t)kSeq * kH * sizeof(float);
    __bf16* Hd    = (__bf16*)ws;  ws += (size_t)kSeq * kH * sizeof(__bf16);
    float*  pmax  = (float*)ws;   ws += (size_t)kSeq * kVSpl * sizeof(float);
    float*  psum  = (float*)ws;   ws += (size_t)kSeq * kVSpl * sizeof(float);
    float*  ptgt  = (float*)ws;   ws += (size_t)kSeq * kVSpl * sizeof(float);
    __bf16* WihB  = (__bf16*)ws;  ws += (size_t)k3H * kH * sizeof(__bf16);
    __bf16* outWB = (__bf16*)ws;  ws += (size_t)kV * kH * sizeof(__bf16);

    // one-time bf16 weight conversions (8 floats per thread)
    f32_to_bf16_kernel<<<(k3H * kH) / (256 * 8), 256, 0, stream>>>(enc_Wih, WihB);
    f32_to_bf16_kernel<<<(kV * kH) / (256 * 8), 256, 0, stream>>>(out_W, outWB);

    gather_emb_kernel<<<(kSeq * kH) / 256, 256, 0, stream>>>(input, enc_emb, Xb);
    dim3 g1(k3H / 512, kSeq / 16);
    gi_gemm_kernel<<<g1, 256, 0, stream>>>(Xb, WihB, enc_bih, Gi);
    enc_seq_kernel<<<1, 1024, 0, stream>>>(enc_Whh, enc_bhh, Gi, encO);
    dec_seq_kernel<<<1, 1024, 0, stream>>>(target, dec_emb, attn_W, attn_b,
                                           comb_W, comb_b, dec_Wih, dec_Whh,
                                           dec_bih, dec_bhh, encO, Hd);
    dim3 g2(kSeq / 16, kVSpl);
    out_nll_partial_kernel<<<g2, 256, 0, stream>>>(Hd, outWB, out_b, target,
                                                   pmax, psum, ptgt);
    nll_finalize_kernel<<<1, 512, 0, stream>>>(pmax, psum, ptgt, out);
}
